// DualWeightAttention_55216099558120
// MI455X (gfx1250) — matrix operand order repaired and checked by reading, not compiled, not run on hardware
//
#include <hip/hip_runtime.h>
#include <hip/hip_bf16.h>

typedef __bf16 bf16_t;
typedef __bf16 v16bf __attribute__((ext_vector_type(16)));
typedef __bf16 v8bf  __attribute__((ext_vector_type(8)));
typedef float  v8f   __attribute__((ext_vector_type(8)));

#define B_  2
#define S_  2048
#define H_  1024
#define NH_ 16
#define HD_ 64
#define M_  (B_ * S_)   // 4096 flattened rows

__device__ __forceinline__ v8f wmma_bf16(v16bf a, v16bf b, v8f c) {
  return __builtin_amdgcn_wmma_f32_16x16x32_bf16(false, a, false, b, (short)0, c, false, false);
}

__device__ __forceinline__ v16bf cat8(v8bf lo, v8bf hi) {
  return __builtin_shufflevector(lo, hi, 0,1,2,3,4,5,6,7,8,9,10,11,12,13,14,15);
}

// ---------------------------------------------------------------------------
// Elementwise: fp32 -> bf16 cast, and (static + dynamic) weight combine
// ---------------------------------------------------------------------------
__global__ void cast_f32_bf16(const float* __restrict__ in, bf16_t* __restrict__ out, int n) {
  int i = blockIdx.x * blockDim.x + threadIdx.x;
  if (i < n) out[i] = (bf16_t)in[i];
}

__global__ void combine_w(const float* __restrict__ a, const float* __restrict__ b,
                          bf16_t* __restrict__ out, int n) {
  int i = blockIdx.x * blockDim.x + threadIdx.x;
  if (i < n) out[i] = (bf16_t)(a[i] + b[i]);
}

// ---------------------------------------------------------------------------
// GEMM: C = A[M,K] * W[N,K]^T  via v_wmma_f32_16x16x32_bf16
// MODE 0: bf16 out [M,N];  MODE 1: bf16 out transposed [N,M];  MODE 2: f32 out [M,N]
// Block = 256 threads (8 waves); wave computes a 16x64 tile. Grid (N/64, M/128).
// ---------------------------------------------------------------------------
template <int MODE>
__global__ __launch_bounds__(256) void gemm_bf16_wmma(
    const bf16_t* __restrict__ A, const bf16_t* __restrict__ W,
    void* __restrict__ Cv, int M, int N, int K) {
  const int lane = threadIdx.x & 31;
  const int wave = threadIdx.x >> 5;
  const int lr   = lane & 15;
  const int hi   = lane >> 4;
  const int m0   = blockIdx.y * 128 + wave * 16;
  const int n0   = blockIdx.x * 64;

  v8f acc[4] = {};
  const bf16_t* arow = A + (size_t)(m0 + lr) * K;

  for (int k0 = 0; k0 < K; k0 += 32) {
    // A fragment 16x32: lanes 0-15 rows, K contiguous {0..15}/{16..31}
    v8bf a_lo = *(const v8bf*)(arow + k0 + hi * 16);
    v8bf a_hi = *(const v8bf*)(arow + k0 + hi * 16 + 8);
    v16bf afrag = cat8(a_lo, a_hi);
#pragma unroll
    for (int nt = 0; nt < 4; ++nt) {
      const bf16_t* wrow = W + (size_t)(n0 + nt * 16 + lr) * K + k0 + hi * 16;
      v16bf bfrag = *(const v16bf*)wrow;  // 16 contiguous bf16 (2 x b128)
      acc[nt] = wmma_bf16(afrag, bfrag, acc[nt]);
    }
  }

#pragma unroll
  for (int nt = 0; nt < 4; ++nt) {
    const int n = n0 + nt * 16 + lr;
    if (MODE == 0) {
      bf16_t* C = (bf16_t*)Cv;
#pragma unroll
      for (int r = 0; r < 8; ++r)
        C[(size_t)(m0 + 8 * hi + r) * N + n] = (bf16_t)acc[nt][r];
    } else if (MODE == 1) {
      bf16_t* C = (bf16_t*)Cv;  // [N, M] — contiguous per-lane 16B store
      v8bf v;
#pragma unroll
      for (int r = 0; r < 8; ++r) v[r] = (bf16_t)acc[nt][r];
      *(v8bf*)(C + (size_t)n * M + m0 + 8 * hi) = v;
    } else {
      float* C = (float*)Cv;
#pragma unroll
      for (int r = 0; r < 8; ++r)
        C[(size_t)(m0 + 8 * hi + r) * N + n] = acc[nt][r];
    }
  }
}

// ---------------------------------------------------------------------------
// Flash attention: one wave per (b, h, 16-row Q block).
// Q,K: bf16 [B*S, H]; Vt: bf16 [H, B*S]; mask fp32 [B,1,S,S]; O: bf16 [B*S, H]
// ---------------------------------------------------------------------------
__global__ __launch_bounds__(256) void attn_fwd(
    const bf16_t* __restrict__ Q, const bf16_t* __restrict__ Km,
    const bf16_t* __restrict__ Vt, const float* __restrict__ mask,
    bf16_t* __restrict__ O) {
  __shared__ bf16_t pshm[8][16 * 32];  // per-wave 16x32 P staging tile

  const int lane = threadIdx.x & 31;
  const int wave = threadIdx.x >> 5;
  const int lr   = lane & 15;
  const int hi   = lane >> 4;
  const int gw   = blockIdx.x * 8 + wave;
  const int qb   = gw & 127;          // S/16 = 128 Q blocks
  const int h    = (gw >> 7) & 15;    // NH = 16
  const int b    = gw >> 11;

  // Q fragments (16 rows x 64 head-dim = 2 A-fragments)
  const bf16_t* Qp = Q + (size_t)(b * S_ + qb * 16 + lr) * H_ + h * HD_;
  v16bf aq[2];
#pragma unroll
  for (int kk = 0; kk < 2; ++kk) {
    v8bf lo = *(const v8bf*)(Qp + kk * 32 + hi * 16);
    v8bf hv = *(const v8bf*)(Qp + kk * 32 + hi * 16 + 8);
    aq[kk] = cat8(lo, hv);
  }

  v8f o0 = {}, o1 = {}, o2 = {}, o3 = {};
  float mrun[8], lrun[8];
#pragma unroll
  for (int r = 0; r < 8; ++r) { mrun[r] = -3.0e38f; lrun[r] = 0.0f; }

  const float* mbase = mask + (size_t)b * S_ * S_;
  const int qrow0 = qb * 16 + 8 * hi;
  bf16_t* pw = &pshm[wave][0];

  for (int kb = 0; kb < S_ / 32; ++kb) {
    // ---- scores: two 16x16 tiles (32 keys), K-dim = 64 via 2 WMMAs each ----
    v8f sc[2];
#pragma unroll
    for (int j = 0; j < 2; ++j) {
      const bf16_t* Kp = Km + (size_t)(b * S_ + kb * 32 + j * 16 + lr) * H_ + h * HD_;
      v16bf bk0 = *(const v16bf*)(Kp + hi * 16);        // d = 0..31
      v16bf bk1 = *(const v16bf*)(Kp + 32 + hi * 16);   // d = 32..63
      v8f s = {};
      s = wmma_bf16(aq[0], bk0, s);
      s = wmma_bf16(aq[1], bk1, s);
      sc[j] = s;
    }
    // ---- scale + additive mask (fp32) ----
#pragma unroll
    for (int j = 0; j < 2; ++j) {
      const int col = kb * 32 + j * 16 + lr;
#pragma unroll
      for (int r = 0; r < 8; ++r)
        sc[j][r] = sc[j][r] * 0.125f + mbase[(size_t)(qrow0 + r) * S_ + col];
    }
    // ---- online softmax: row max / exp / row sum (16-lane reductions) ----
    float mnew[8], fac[8];
#pragma unroll
    for (int r = 0; r < 8; ++r) {
      float mx = fmaxf(sc[0][r], sc[1][r]);
#pragma unroll
      for (int off = 8; off >= 1; off >>= 1)
        mx = fmaxf(mx, __shfl_xor(mx, off, 16));
      mnew[r] = fmaxf(mrun[r], mx);
      fac[r]  = __expf(mrun[r] - mnew[r]);
    }
#pragma unroll
    for (int r = 0; r < 8; ++r) {
      float p0 = __expf(sc[0][r] - mnew[r]);
      float p1 = __expf(sc[1][r] - mnew[r]);
      sc[0][r] = p0; sc[1][r] = p1;
      float rs = p0 + p1;
#pragma unroll
      for (int off = 8; off >= 1; off >>= 1)
        rs += __shfl_xor(rs, off, 16);
      lrun[r] = lrun[r] * fac[r] + rs;
      mrun[r] = mnew[r];
      o0[r] *= fac[r]; o1[r] *= fac[r]; o2[r] *= fac[r]; o3[r] *= fac[r];
    }
    // ---- P: C-layout -> A-layout via wave-private LDS tile ----
#pragma unroll
    for (int j = 0; j < 2; ++j)
#pragma unroll
      for (int r = 0; r < 8; ++r)
        pw[(r + 8 * hi) * 32 + j * 16 + lr] = (bf16_t)sc[j][r];
    // same-wave DS ops are in-order; compiler inserts s_wait_dscnt before use
    v8bf plo = *(const v8bf*)(pw + lr * 32 + hi * 16);
    v8bf phv = *(const v8bf*)(pw + lr * 32 + hi * 16 + 8);
    v16bf pf = cat8(plo, phv);
    // ---- PV: out(16x64) += P(16x32) @ V(32x64), V transposed => contiguous ----
    const size_t vcol = (size_t)b * S_ + kb * 32 + hi * 16;
    o0 = wmma_bf16(pf, *(const v16bf*)(Vt + (size_t)(h * HD_ +  0 + lr) * M_ + vcol), o0);
    o1 = wmma_bf16(pf, *(const v16bf*)(Vt + (size_t)(h * HD_ + 16 + lr) * M_ + vcol), o1);
    o2 = wmma_bf16(pf, *(const v16bf*)(Vt + (size_t)(h * HD_ + 32 + lr) * M_ + vcol), o2);
    o3 = wmma_bf16(pf, *(const v16bf*)(Vt + (size_t)(h * HD_ + 48 + lr) * M_ + vcol), o3);
  }

  // ---- normalize by running sum and store bf16 [B*S, H] ----
  bf16_t* Op = O + (size_t)(b * S_) * H_ + h * HD_;
#pragma unroll
  for (int r = 0; r < 8; ++r) {
    const float inv = 1.0f / lrun[r];
    const int row = qb * 16 + 8 * hi + r;
    Op[(size_t)row * H_ +  0 + lr] = (bf16_t)(o0[r] * inv);
    Op[(size_t)row * H_ + 16 + lr] = (bf16_t)(o1[r] * inv);
    Op[(size_t)row * H_ + 32 + lr] = (bf16_t)(o2[r] * inv);
    Op[(size_t)row * H_ + 48 + lr] = (bf16_t)(o3[r] * inv);
  }
}

// ---------------------------------------------------------------------------
extern "C" void kernel_launch(void* const* d_in, const int* in_sizes, int n_in,
                              void* d_out, int out_size, void* d_ws, size_t ws_size,
                              hipStream_t stream) {
  const float* x   = (const float*)d_in[0];
  const float* msk = (const float*)d_in[1];
  const float* qs  = (const float*)d_in[2];
  const float* qd  = (const float*)d_in[3];
  const float* ks  = (const float*)d_in[4];
  const float* kd  = (const float*)d_in[5];
  const float* vs  = (const float*)d_in[6];
  const float* vd  = (const float*)d_in[7];
  const float* os_ = (const float*)d_in[8];
  const float* od  = (const float*)d_in[9];

  // workspace layout (48 MB total)
  char* ws = (char*)d_ws;
  bf16_t* xb  = (bf16_t*)(ws);                    // [4096,1024] bf16   8 MB
  bf16_t* wq  = (bf16_t*)(ws + ( 8ull << 20));    // [1024,1024] bf16   2 MB
  bf16_t* wk  = (bf16_t*)(ws + (10ull << 20));
  bf16_t* wv  = (bf16_t*)(ws + (12ull << 20));
  bf16_t* wo  = (bf16_t*)(ws + (14ull << 20));
  bf16_t* Qb  = (bf16_t*)(ws + (16ull << 20));    // [4096,1024] bf16   8 MB
  bf16_t* Kb  = (bf16_t*)(ws + (24ull << 20));
  bf16_t* Vtb = (bf16_t*)(ws + (32ull << 20));    // transposed [1024,4096]
  bf16_t* Ab  = (bf16_t*)(ws + (40ull << 20));    // attention output bf16

  const int nx = M_ * H_;  // 4M
  const int nw = H_ * H_;  // 1M
  cast_f32_bf16<<<nx / 256, 256, 0, stream>>>(x, xb, nx);
  combine_w<<<nw / 256, 256, 0, stream>>>(qs, qd, wq, nw);
  combine_w<<<nw / 256, 256, 0, stream>>>(ks, kd, wk, nw);
  combine_w<<<nw / 256, 256, 0, stream>>>(vs, vd, wv, nw);
  combine_w<<<nw / 256, 256, 0, stream>>>(os_, od, wo, nw);

  dim3 gg(H_ / 64, M_ / 128);  // (16, 32)
  gemm_bf16_wmma<0><<<gg, 256, 0, stream>>>(xb, wq, (void*)Qb,  M_, H_, H_);
  gemm_bf16_wmma<0><<<gg, 256, 0, stream>>>(xb, wk, (void*)Kb,  M_, H_, H_);
  gemm_bf16_wmma<1><<<gg, 256, 0, stream>>>(xb, wv, (void*)Vtb, M_, H_, H_);

  // B*NH*(S/16) = 4096 waves, 8 per block
  attn_fwd<<<512, 256, 0, stream>>>(Qb, Kb, Vtb, msk, Ab);

  gemm_bf16_wmma<2><<<gg, 256, 0, stream>>>(Ab, wo, d_out, M_, H_, H_);
}
